// GCNEncoder_55052890800619
// MI455X (gfx1250) — matrix operand order repaired and checked by reading, not compiled, not run on hardware
//
#include <hip/hip_runtime.h>

// ---------- types ----------
typedef __bf16 bf16_t;
typedef bf16_t        v16bf  __attribute__((ext_vector_type(16)));
typedef float         v8f    __attribute__((ext_vector_type(8)));
typedef unsigned short v8u16  __attribute__((ext_vector_type(8)));
typedef unsigned short v16u16 __attribute__((ext_vector_type(16)));

union BF16Vec { v16u16 u; v16bf b; };

__device__ __forceinline__ unsigned short f2bf(float f) {
    unsigned u = __float_as_uint(f);
    u += 0x7FFFu + ((u >> 16) & 1u);          // round-to-nearest-even
    return (unsigned short)(u >> 16);
}

// ---------- degree / dinv ----------
__global__ __launch_bounds__(256) void deg_kernel(const long long* __restrict__ dst,
                                                  float* __restrict__ deg, long long E) {
    long long e = (long long)blockIdx.x * 256 + threadIdx.x;
    if (e < E) atomicAdd(&deg[(int)dst[e]], 1.0f);
}

__global__ __launch_bounds__(256) void dinv_kernel(float* __restrict__ d, int N) {
    int i = blockIdx.x * 256 + threadIdx.x;
    if (i < N) d[i] = rsqrtf(d[i] + 1.0f);    // +1 for the self loop
}

// ---------- pack W (row-major f32 K x NC) into WMMA B-operand bf16 layout ----------
// B 32x16 bf16 per WMMA: lanes 0-15 hold K=0..15 of column N=lane, lanes 16-31 hold
// K=16..31 of column N=lane-16; 16 bf16 per lane = 32B. One (kstep,ntile) block = 1KB.
__global__ __launch_bounds__(256) void pack_w(const float* __restrict__ W,
                                              unsigned short* __restrict__ wp,
                                              int K, int NC) {
    int idx = blockIdx.x * 256 + threadIdx.x;
    if (idx >= K * NC) return;
    int k = idx / NC, n = idx % NC;
    int ntile = n >> 4;
    int kstep = k >> 5;
    int lane  = (n & 15) + (((k & 31) >> 4) << 4);
    int elem  = k & 15;
    size_t off = ((size_t)(ntile * (K / 32) + kstep) * 512) + (size_t)lane * 16 + elem;
    wp[off] = f2bf(W[idx]);
}

// ---------- WMMA GEMM: C[N x NC] = A[N x K] * Wp ----------
template <int K, int NC>
__global__ __launch_bounds__(256) void gemm_wmma(const float* __restrict__ A,
                                                 const unsigned short* __restrict__ Wp,
                                                 float* __restrict__ C, int Nrows) {
    constexpr int STR = K + 8;                 // bf16 units; +16B pad -> conflict-free b128 reads
    __shared__ unsigned short sA[32 * STR];
    const int t  = threadIdx.x;
    const int m0 = blockIdx.x * 32;

    // stage A tile (32 x K) into LDS as bf16, row-major, padded stride
    constexpr int PAIRS = 16 * K;              // 32*K/2 float2 pairs
    for (int p = t; p < PAIRS; p += 256) {
        const int row = p / (K / 2);
        const int kp  = p % (K / 2);
        const int gr  = m0 + row;
        unsigned pk = 0;
        if (gr < Nrows) {
            const float2 f = *(const float2*)(A + (size_t)gr * K + (size_t)kp * 2);
            pk = (unsigned)f2bf(f.x) | ((unsigned)f2bf(f.y) << 16);
        }
        *(unsigned*)&sA[row * STR + kp * 2] = pk;
    }
    __syncthreads();

    const int lane  = t & 31;
    const int w     = t >> 5;                  // 8 waves: 2 M-subtiles x 4 N-subtiles
    const int mrow  = (w >> 2) * 16;
    const int ntile = blockIdx.y * 4 + (w & 3);
    const unsigned short* wpB = Wp + (size_t)ntile * (K / 32) * 512 + (size_t)lane * 16;

    // A operand k order used here (same as packed B): lane<16 -> K {0..15}, lane>=16 -> K {16..31}
    const int arow = mrow + (lane & 15);
    const int koff = (lane < 16) ? 0 : 16;
    const unsigned short* pa0 = &sA[arow * STR + koff];

    v8f acc = {0.f, 0.f, 0.f, 0.f, 0.f, 0.f, 0.f, 0.f};
#pragma unroll
    for (int kk = 0; kk < K / 32; ++kk) {
        v8u16 a0  = *(const v8u16*)(pa0 + kk * 32);
        v8u16 a1  = *(const v8u16*)(pa0 + kk * 32 + 8);
        v8u16 bb0 = *(const v8u16*)(wpB + kk * 512);
        v8u16 bb1 = *(const v8u16*)(wpB + kk * 512 + 8);
        BF16Vec av, bv;
        av.u = __builtin_shufflevector(a0, a1, 0,1,2,3,4,5,6,7,8,9,10,11,12,13,14,15);
        bv.u = __builtin_shufflevector(bb0, bb1, 0,1,2,3,4,5,6,7,8,9,10,11,12,13,14,15);
        acc = __builtin_amdgcn_wmma_f32_16x16x32_bf16(false, av.b, false, bv.b,
                                                      (short)0, acc, false, false);
    }

    // C/D layout: VGPR g -> M = g (lanes 0-15) or 8+g (lanes 16-31), N = lane&15
    const int rbase = m0 + mrow + ((lane >= 16) ? 8 : 0);
    const int col   = ntile * 16 + (lane & 15);
    float* __restrict__ cp = C + (size_t)rbase * NC + col;
    if (m0 + 32 <= Nrows) {
        // full tile: unconditional stores, constant row-stride immediate offsets
#pragma unroll
        for (int g = 0; g < 8; ++g) cp[(size_t)g * NC] = acc[g];
    } else {
#pragma unroll
        for (int g = 0; g < 8; ++g) {
            if (rbase + g < Nrows) cp[(size_t)g * NC] = acc[g];
        }
    }
}

// ---------- edge scatter: agg[dst] += xw[src] * dinv[src]*dinv[dst] ----------
template <int Hc>
__global__ __launch_bounds__(256) void scatter_kernel(const float* __restrict__ xw,
                                                      const long long* __restrict__ src,
                                                      const long long* __restrict__ dst,
                                                      const float* __restrict__ dinv,
                                                      float* __restrict__ agg, long long E) {
    constexpr int CG = Hc / 4;                 // float4 chunks per row
    long long gid = (long long)blockIdx.x * 256 + threadIdx.x;
    long long e   = gid / CG;
    if (e >= E) return;
    int c = (int)(gid - e * CG);
    int s = (int)src[e], d = (int)dst[e];
    float nm = dinv[s] * dinv[d];
    const float4 v = ((const float4*)(xw + (size_t)s * Hc))[c];
    float* o = agg + (size_t)d * Hc + (size_t)c * 4;
    atomicAdd(o + 0, v.x * nm);
    atomicAdd(o + 1, v.y * nm);
    atomicAdd(o + 2, v.z * nm);
    atomicAdd(o + 3, v.w * nm);
}

// ---------- finalize: io = [relu](agg + xw*dinv^2 + bias) ----------
template <int Hc, bool RELU>
__global__ __launch_bounds__(256) void finalize_kernel(float* __restrict__ io,
                                                       const float* __restrict__ xw,
                                                       const float* __restrict__ dinv,
                                                       const float* __restrict__ bias,
                                                       long long total) {
    long long t = (long long)blockIdx.x * 256 + threadIdx.x;
    if (t >= total) return;
    int i = (int)(t / Hc);
    int j = (int)(t % Hc);
    float di = dinv[i];
    float v  = io[t] + xw[t] * di * di + bias[j];
    io[t] = RELU ? fmaxf(v, 0.f) : v;
}

// ---------- launch ----------
extern "C" void kernel_launch(void* const* d_in, const int* in_sizes, int n_in,
                              void* d_out, int out_size, void* d_ws, size_t ws_size,
                              hipStream_t stream) {
    const float* emb = (const float*)d_in[0];
    const float* W1  = (const float*)d_in[1];
    const float* b1  = (const float*)d_in[2];
    const float* W2  = (const float*)d_in[3];
    const float* b2  = (const float*)d_in[4];
    const long long* ei = (const long long*)d_in[5];   // int64 edge_index [2,E]

    const int D  = in_sizes[4];          // 128
    const int H  = in_sizes[2];          // 256
    const int N  = in_sizes[0] / D;      // 100000
    const long long E = (long long)in_sizes[5] / 2;
    const long long* src = ei;
    const long long* dst = ei + E;

    auto alignup = [](size_t x) { return (x + 255) & ~(size_t)255; };
    char* ws = (char*)d_ws;
    size_t off = 0;
    float* dinv = (float*)(ws + off);          off += alignup((size_t)N * 4);
    unsigned short* wp1 = (unsigned short*)(ws + off); off += alignup((size_t)D * H * 2);
    unsigned short* wp2 = (unsigned short*)(ws + off); off += alignup((size_t)H * D * 2);
    float* bufA = (float*)(ws + off);          off += (size_t)N * H * 4;   // xw (layer1: NxH, layer2: NxD)
    float* bufB = (float*)(ws + off);          off += (size_t)N * H * 4;   // agg1 -> x1 (in place)
    (void)ws_size; (void)n_in;

    // degrees + dinv
    hipMemsetAsync(dinv, 0, (size_t)N * 4, stream);
    deg_kernel<<<(unsigned)((E + 255) / 256), 256, 0, stream>>>(dst, dinv, E);
    dinv_kernel<<<(N + 255) / 256, 256, 0, stream>>>(dinv, N);

    // pack weights into WMMA B layout
    pack_w<<<(D * H + 255) / 256, 256, 0, stream>>>(W1, wp1, D, H);
    pack_w<<<(H * D + 255) / 256, 256, 0, stream>>>(W2, wp2, H, D);

    // layer 1: xw1 = emb @ W1   (K=128, NC=256)
    gemm_wmma<128, 256><<<dim3((N + 31) / 32, 256 / 64), 256, 0, stream>>>(emb, wp1, bufA, N);
    hipMemsetAsync(bufB, 0, (size_t)N * H * 4, stream);
    {
        unsigned long long tt = (unsigned long long)E * (256 / 4);
        scatter_kernel<256><<<(unsigned)((tt + 255) / 256), 256, 0, stream>>>(bufA, src, dst, dinv, bufB, E);
    }
    finalize_kernel<256, true><<<(unsigned)(((long long)N * 256 + 255) / 256), 256, 0, stream>>>(
        bufB, bufA, dinv, b1, (long long)N * 256);

    // layer 2: xw2 = x1 @ W2   (K=256, NC=128)
    gemm_wmma<256, 128><<<dim3((N + 31) / 32, 128 / 64), 256, 0, stream>>>(bufB, wp2, bufA, N);
    hipMemsetAsync(d_out, 0, (size_t)out_size * sizeof(float), stream);
    {
        unsigned long long tt = (unsigned long long)E * (128 / 4);
        scatter_kernel<128><<<(unsigned)((tt + 255) / 256), 256, 0, stream>>>(bufA, src, dst, dinv,
                                                                              (float*)d_out, E);
    }
    finalize_kernel<128, false><<<(unsigned)(((long long)N * 128 + 255) / 256), 256, 0, stream>>>(
        (float*)d_out, bufA, dinv, b2, (long long)N * 128);
}
